// EMAVectorQuantizer_6253472383348
// MI455X (gfx1250) — matrix operand order repaired and checked by reading, not compiled, not run on hardware
//
#include <hip/hip_runtime.h>

typedef __attribute__((ext_vector_type(16))) _Float16 v16h;
typedef __attribute__((ext_vector_type(8)))  float    v8f;
typedef __attribute__((ext_vector_type(4)))  float    v4f;

// Problem shape
#define NROWS   32768      // 32 * 32 * 32 spatial positions
#define DIM     256        // channels / embedding dim
#define KCODES  8192       // codebook size
#define HW      1024       // 32*32
#define NELEM   8388608    // 32*256*32*32
// ws layout (bytes)
#define WSA_OFF 0u                 // A fragments: 32768*256 halves = 16 MB
#define WSB_OFF (16u << 20)        // B fragments: 8192*256 halves = 4 MB
#define EN2_OFF (20u << 20)        // ||e||^2 : 8192 f32 = 32 KB
#define SUM_OFF ((20u << 20) + (64u << 10))

// ---------------------------------------------------------------------------
// Pack z (NCHW f32) into WMMA A-fragment layout, f16.
// wsA element index: ((g*8 + kc)*32 + lane)*16 + h
// A k order chosen here (contiguous, matches B): lane m=L%16; half h <-> k = (L/16)*16 + h
// zf[n][k] = z[b=n/1024][k][pos=n%1024]
// ---------------------------------------------------------------------------
__global__ __launch_bounds__(256) void pack_z_kernel(const float* __restrict__ z,
                                                     _Float16* __restrict__ wsA) {
    int tid  = blockIdx.x * 256 + threadIdx.x;   // 2048 groups * 8 kc * 32 lanes
    int lane = tid & 31;
    int kc   = (tid >> 5) & 7;
    int g    = tid >> 8;
    int m    = lane & 15;
    int n    = g * 16 + m;
    int b    = n >> 10;
    int pos  = n & 1023;
    const float* zb = z + (size_t)b * (DIM * HW) + pos;
    v16h frag;
#pragma unroll
    for (int h = 0; h < 16; ++h) {
        int klocal = ((lane >> 4) << 4) + h;
        int k      = kc * 32 + klocal;
        frag[h]    = (_Float16)zb[(size_t)k * HW];
    }
    *(v16h*)(wsA + (size_t)tid * 16) = frag;
}

// ---------------------------------------------------------------------------
// Pack embedding (row-major f32 [8192,256]) into WMMA B-fragment layout, f16.
// B k order chosen here (contiguous, matches A): lane n=L%16; half h <-> k = (L/16)*16 + h
// ---------------------------------------------------------------------------
__global__ __launch_bounds__(256) void pack_emb_kernel(const float* __restrict__ emb,
                                                       _Float16* __restrict__ wsB) {
    int tid   = blockIdx.x * 256 + threadIdx.x;  // 512 tiles * 8 kc * 32 lanes
    int lane  = tid & 31;
    int kc    = (tid >> 5) & 7;
    int t     = tid >> 8;
    int code  = t * 16 + (lane & 15);
    int kbase = kc * 32 + ((lane >> 4) << 4);
    const float* src = emb + (size_t)code * DIM + kbase;
    v16h frag;
#pragma unroll
    for (int h = 0; h < 16; ++h) frag[h] = (_Float16)src[h];
    *(v16h*)(wsB + (size_t)tid * 16) = frag;
}

// ---------------------------------------------------------------------------
// ||e||^2 per code (fp32, exact)
// ---------------------------------------------------------------------------
__global__ __launch_bounds__(256) void emb_norm_kernel(const float* __restrict__ emb,
                                                       float* __restrict__ en2) {
    int c = blockIdx.x * 256 + threadIdx.x;      // 8192 codes
    const v4f* row = (const v4f*)(emb + (size_t)c * DIM);
    float s = 0.0f;
#pragma unroll 8
    for (int i = 0; i < DIM / 4; ++i) {
        v4f v = row[i];
        s += v.x * v.x + v.y * v.y + v.z * v.z + v.w * v.w;
    }
    en2[c] = s;
}

__global__ void zero_kernel(float* p) { *p = 0.0f; }

// ---------------------------------------------------------------------------
// Fused distance-GEMM + argmin.
// 8 waves/WG; each wave owns M=32 z-rows (two row-groups, A resident in
// 128 VGPRs for full d=256), streams 512 tiles of 16 codes. Each loaded B
// fragment feeds TWO independent WMMA accumulation chains -> half the
// loads/waits per WMMA and half the codebook re-streaming (128 WGs).
// dist = ||e||^2 - 2*(z.e); ||z||^2 dropped (row-constant).
// Per-lane running (min,argmin): N is striped across lanes in the C layout,
// so each lane covers a fixed code residue -> no cross-lane work until one
// butterfly per row at the end.
// ---------------------------------------------------------------------------
__global__ __launch_bounds__(256) void argmin_kernel(const _Float16* __restrict__ wsA,
                                                     const _Float16* __restrict__ wsB,
                                                     const float* __restrict__ en2,
                                                     int* __restrict__ idx_out) {
    int lane  = threadIdx.x & 31;
    int wave  = threadIdx.x >> 5;
    int G     = blockIdx.x * 8 + wave;           // 32-row super-group
    int g0    = G * 2;                           // row groups g0, g0+1
    int nlane = lane & 15;

    v16h a0[8], a1[8];
#pragma unroll
    for (int kc = 0; kc < 8; ++kc) {
        a0[kc] = ((const v16h*)wsA)[((size_t)g0 * 8 + kc) * 32 + lane];
        a1[kc] = ((const v16h*)wsA)[((size_t)(g0 + 1) * 8 + kc) * 32 + lane];
    }

    float best0[8], best1[8];
    int   bidx0[8], bidx1[8];
#pragma unroll
    for (int r = 0; r < 8; ++r) {
        best0[r] = 3.4e38f; bidx0[r] = 0;
        best1[r] = 3.4e38f; bidx1[r] = 0;
    }

    const v16h* bBase = (const v16h*)wsB + lane;
    for (int t = 0; t < 512; ++t) {
        float en2v = en2[t * 16 + nlane];
        const v16h* bPtr = bBase + (size_t)t * 256;
        __builtin_prefetch(bPtr + 256, 0, 1);    // next code tile
        v8f acc0 = {};
        v8f acc1 = {};
#pragma unroll
        for (int kc = 0; kc < 8; ++kc) {
            v16h bf = bPtr[kc * 32];
            acc0 = __builtin_amdgcn_wmma_f32_16x16x32_f16(
                false, a0[kc], false, bf, (short)0, acc0, false, false);
            acc1 = __builtin_amdgcn_wmma_f32_16x16x32_f16(
                false, a1[kc], false, bf, (short)0, acc1, false, false);
        }
        int code = t * 16 + nlane;
#pragma unroll
        for (int r = 0; r < 8; ++r) {
            float d0 = en2v - 2.0f * acc0[r];
            float d1 = en2v - 2.0f * acc1[r];
            if (d0 < best0[r]) { best0[r] = d0; bidx0[r] = code; }
            if (d1 < best1[r]) { best1[r] = d1; bidx1[r] = code; }
        }
    }

    // reduce over the 16 lanes of each half (masks < 16 never cross halves)
#pragma unroll
    for (int r = 0; r < 8; ++r) {
        float bv0 = best0[r]; int bi0 = bidx0[r];
        float bv1 = best1[r]; int bi1 = bidx1[r];
#pragma unroll
        for (int mask = 8; mask >= 1; mask >>= 1) {
            float ov0 = __shfl_xor(bv0, mask, 32);
            int   oi0 = __shfl_xor(bi0, mask, 32);
            float ov1 = __shfl_xor(bv1, mask, 32);
            int   oi1 = __shfl_xor(bi1, mask, 32);
            if (ov0 < bv0 || (ov0 == bv0 && oi0 < bi0)) { bv0 = ov0; bi0 = oi0; }
            if (ov1 < bv1 || (ov1 == bv1 && oi1 < bi1)) { bv1 = ov1; bi1 = oi1; }
        }
        if (nlane == 0) {
            int m = r + ((lane >> 4) << 3);      // lanes 0-15 -> rows 0-7, 16-31 -> 8-15
            idx_out[g0 * 16 + m]       = bi0;
            idx_out[(g0 + 1) * 16 + m] = bi1;
        }
    }
}

// ---------------------------------------------------------------------------
// z_q = E[idx] written in NCHW + sum of (z_q - z)^2 via LDS tree + atomic
// ---------------------------------------------------------------------------
__global__ __launch_bounds__(256) void gather_kernel(const float* __restrict__ z,
                                                     const float* __restrict__ emb,
                                                     const int* __restrict__ idx,
                                                     float* __restrict__ zq_out,
                                                     float* __restrict__ sum_accum) {
    __shared__ float sred[256];
    size_t tid = (size_t)blockIdx.x * 256 + threadIdx.x;  // 8388608 elements
    int pos = (int)(tid & 1023);
    int ch  = (int)((tid >> 10) & 255);
    int b   = (int)(tid >> 18);
    int code = idx[b * HW + pos];
    float e  = emb[(size_t)code * DIM + ch];
    float zv = z[tid];
    zq_out[tid] = e;
    float d = e - zv;
    sred[threadIdx.x] = d * d;
    __syncthreads();
#pragma unroll
    for (int s = 128; s > 0; s >>= 1) {
        if ((int)threadIdx.x < s) sred[threadIdx.x] += sred[threadIdx.x + s];
        __syncthreads();
    }
    if (threadIdx.x == 0) atomicAdd(sum_accum, sred[0]);
}

__global__ void finalize_kernel(const float* __restrict__ sum_accum,
                                float* __restrict__ diff_out) {
    *diff_out = 0.25f * (*sum_accum) / (float)NELEM;
}

// ---------------------------------------------------------------------------
extern "C" void kernel_launch(void* const* d_in, const int* in_sizes, int n_in,
                              void* d_out, int out_size, void* d_ws, size_t ws_size,
                              hipStream_t stream) {
    const float* z   = (const float*)d_in[0];
    const float* emb = (const float*)d_in[1];

    float* out     = (float*)d_out;
    float* zq      = out;                        // 8,388,608 f32
    float* diff    = out + NELEM;                // 1 f32
    int*   idx_out = (int*)(out + NELEM + 1);    // 32768 int32

    char* ws = (char*)d_ws;
    _Float16* wsA  = (_Float16*)(ws + WSA_OFF);
    _Float16* wsB  = (_Float16*)(ws + WSB_OFF);
    float*    en2  = (float*)(ws + EN2_OFF);
    float*    ssum = (float*)(ws + SUM_OFF);

    pack_z_kernel  <<<(2048 * 8 * 32) / 256, 256, 0, stream>>>(z, wsA);
    pack_emb_kernel<<<(512 * 8 * 32) / 256,  256, 0, stream>>>(emb, wsB);
    emb_norm_kernel<<<KCODES / 256,          256, 0, stream>>>(emb, en2);
    zero_kernel    <<<1, 1, 0, stream>>>(ssum);
    argmin_kernel  <<<128,   256, 0, stream>>>(wsA, wsB, en2, idx_out);
    gather_kernel  <<<NELEM / 256, 256, 0, stream>>>(z, emb, idx_out, zq, ssum);
    finalize_kernel<<<1, 1, 0, stream>>>(ssum, diff);
}
